// AttnMoveModel_59382217834695
// MI455X (gfx1250) — matrix operand order repaired and checked by reading, not compiled, not run on hardware
//
#include <hip/hip_runtime.h>
#include <math.h>

typedef __attribute__((ext_vector_type(16))) __bf16 v16bf;
typedef __attribute__((ext_vector_type(8)))  __bf16 v8bf;
typedef __attribute__((ext_vector_type(8)))  float  v8f;
typedef __attribute__((ext_vector_type(4)))  int    v4i;

#define D_MODEL 512
#define B_SZ    32
#define S_CURR  128
#define N_MASKP 16
#define HEADS   8
#define N_CAND  39998

#define AS1 __attribute__((address_space(1)))
#define AS3 __attribute__((address_space(3)))

#if __has_builtin(__builtin_amdgcn_global_load_async_to_lds_b128)
#define HAVE_ASYNC_LDS 1
#endif

// ---------------- helpers ----------------
__device__ __forceinline__ v8f wmma_bf16(v16bf a, v16bf b, v8f c) {
  return __builtin_amdgcn_wmma_f32_16x16x32_bf16(false, a, false, b, (short)0, c, false, false);
}

__device__ __forceinline__ v16bf cat16(v8bf a, v8bf b) {
  return __builtin_shufflevector(a, b, 0, 1, 2, 3, 4, 5, 6, 7, 8, 9, 10, 11, 12, 13, 14, 15);
}

// A-fragment (16-bit, 16x32): lane half 'hi': k = k0 + hi*16 + e  (contiguous 16)
__device__ __forceinline__ v16bf loadA(const __bf16* row, int k0, int hi) {
  v8bf a = *reinterpret_cast<const v8bf*>(row + k0 + hi * 16);
  v8bf b = *reinterpret_cast<const v8bf*>(row + k0 + hi * 16 + 8);
  return cat16(a, b);
}

// B-fragment (16-bit, 32x16): lane half 'hi': k = k0 + hi*16 + e  (contiguous 16)
__device__ __forceinline__ v16bf loadB(const __bf16* row, int k0, int hi) {
  v8bf a = *reinterpret_cast<const v8bf*>(row + k0 + hi * 16);
  v8bf b = *reinterpret_cast<const v8bf*>(row + k0 + hi * 16 + 8);
  return cat16(a, b);
}

// fp32 -> bf16 hi + lo (native v_cvt_pk_bf16_f32 path, RNE)
__device__ __forceinline__ void split1(float f, __bf16& h, __bf16& l) {
  __bf16 hb = (__bf16)f;
  float hf = (float)hb;
  h = hb;
  l = (__bf16)(f - hf);
}

__device__ __forceinline__ void splitf16(const float* v, v16bf& h, v16bf& l) {
#pragma unroll
  for (int i = 0; i < 16; ++i) {
    __bf16 hb, lb;
    split1(v[i], hb, lb);
    h[i] = hb;
    l[i] = lb;
  }
}

// triple-WMMA near-fp32 step: ah*bh + ah*bl + al*bh
__device__ __forceinline__ v8f mm3s(v16bf ah, v16bf al, v16bf bh, v16bf bl, v8f acc) {
  acc = wmma_bf16(ah, bh, acc);
  acc = wmma_bf16(ah, bl, acc);
  acc = wmma_bf16(al, bh, acc);
  return acc;
}

__device__ __forceinline__ void wait_async_all() {
#if __has_builtin(__builtin_amdgcn_s_wait_asynccnt)
  __builtin_amdgcn_s_wait_asynccnt(0);
#else
  asm volatile("s_wait_asynccnt 0x0" ::: "memory");
#endif
}

// ---------- curr = emb[mask_curr] + positional embedding ----------
__global__ void build_curr_kernel(const int* __restrict__ idx, const float* __restrict__ emb,
                                  float* __restrict__ curr) {
  int s = blockIdx.x;   // 0..127
  int b = blockIdx.y;   // 0..31
  int t = threadIdx.x;  // 0..127, 4 floats each
  int row = idx[b * S_CURR + s];
  float4 e = reinterpret_cast<const float4*>(emb + (size_t)row * D_MODEL)[t];
  const float cc = -0.017988946039015984f;  // -ln(10000)/512
  float sp = (float)s;
  int d0 = t * 4;  // even
  float w0 = __expf(cc * (float)d0);
  float w2 = __expf(cc * (float)(d0 + 2));
  float4 o;
  o.x = e.x + sinf(sp * w0);
  o.y = e.y + cosf(sp * w0);
  o.z = e.z + sinf(sp * w2);
  o.w = e.w + cosf(sp * w2);
  reinterpret_cast<float4*>(curr + (size_t)(b * S_CURR + s) * D_MODEL)[t] = o;
}

// ---------- elementwise split (optionally tanh first) ----------
__global__ void split_kernel(const float* __restrict__ src, __bf16* __restrict__ hi,
                             __bf16* __restrict__ lo, int n, int do_tanh) {
  int i = blockIdx.x * 256 + threadIdx.x;
  if (i < n) {
    float f = src[i];
    if (do_tanh) f = tanhf(f);
    __bf16 hb, lb;
    split1(f, hb, lb);
    hi[i] = hb;
    lo[i] = lb;
  }
}

// ---------- gather masked query rows (fp32) ----------
__global__ void qgather_kernel(const int* __restrict__ mask_pos, const float* __restrict__ curr,
                               float* __restrict__ qin) {
  int m = blockIdx.x & 15;
  int b = blockIdx.x >> 4;
  int t = threadIdx.x;
  int pos = mask_pos[b * N_MASKP + m];
  reinterpret_cast<float4*>(qin + (size_t)(b * N_MASKP + m) * D_MODEL)[t] =
      reinterpret_cast<const float4*>(curr + (size_t)(b * S_CURR + pos) * D_MODEL)[t];
}

// ---------- GEMM: Y[M,N] = X[M,K] @ W[N,K]^T (+bias) ----------
// X pre-split bf16 hi/lo planes; W fp32 staged once per block into LDS via
// global_load_async_to_lds (all 4 waves share the 16-column strip), then
// converted per 16-col fragment amortized over 8 m-tiles.
// One wave = 128 rows x 16 cols. blockDim=128 (4 waves). grid = (ceil(N/16), M/512).
// Output: Yf (fp32) and/or Yhi/Ylo (bf16 split planes).
__global__ void gemm8_kernel(const __bf16* __restrict__ Xhi, const __bf16* __restrict__ Xlo,
                             const float* __restrict__ W, const float* __restrict__ bias,
                             float* __restrict__ Yf, __bf16* __restrict__ Yhi,
                             __bf16* __restrict__ Ylo, int M, int N, int K) {
  __shared__ float Bs[16 * 512];  // 32 KB: 16 columns x K (<=512) fp32
  int lane = threadIdx.x & 31;
  int wave = threadIdx.x >> 5;
  int L2 = lane & 15, hi = lane >> 4;
  int rbase = (blockIdx.y * 4 + wave) * 128;
  int col = blockIdx.x * 16 + L2;
  bool colok = col < N;

  // ---- cooperative stage of the B strip (16 cols x K) into LDS ----
  {
    int segs_per_col = K >> 2;      // 16-byte segments per column
    int total = 16 * segs_per_col;  // 2048 for K=512
    for (int seg = threadIdx.x; seg < total; seg += 128) {
      int c = seg / segs_per_col;
      int ko = (seg - c * segs_per_col) * 4;
      int gcol = blockIdx.x * 16 + c;
      if (gcol >= N) gcol = 0;  // clamp; zeroed at convert time via colok
      const float* src = W + (size_t)gcol * K + ko;
      float* dst = Bs + c * K + ko;
#ifdef HAVE_ASYNC_LDS
      __builtin_amdgcn_global_load_async_to_lds_b128((AS1 v4i*)src, (AS3 v4i*)dst, 0, 0);
#else
      *reinterpret_cast<float4*>(dst) = *reinterpret_cast<const float4*>(src);
#endif
    }
#ifdef HAVE_ASYNC_LDS
    wait_async_all();
#endif
    __syncthreads();
  }

  v8f acc[8];
  v8f z = {};
#pragma unroll
  for (int mt = 0; mt < 8; ++mt) acc[mt] = z;

  const float* brow = Bs + L2 * K;  // this lane's column, resident in LDS
  for (int k0 = 0; k0 < K; k0 += 32) {
    // B fragment: 16 fp32 from LDS (k = k0 + hi*16 + e), convert once, reuse over 8 m-tiles
    float bv[16];
    const float4* q0 = reinterpret_cast<const float4*>(brow + k0 + hi * 16);
    float4 b0 = q0[0], b1 = q0[1], b2 = q0[2], b3 = q0[3];
    bv[0] = b0.x; bv[1] = b0.y; bv[2] = b0.z; bv[3] = b0.w;
    bv[4] = b1.x; bv[5] = b1.y; bv[6] = b1.z; bv[7] = b1.w;
    bv[8] = b2.x; bv[9] = b2.y; bv[10] = b2.z; bv[11] = b2.w;
    bv[12] = b3.x; bv[13] = b3.y; bv[14] = b3.z; bv[15] = b3.w;
    if (!colok) {
#pragma unroll
      for (int e = 0; e < 16; ++e) bv[e] = 0.f;
    }
    v16bf bh, bl;
    splitf16(bv, bh, bl);
#pragma unroll
    for (int mt = 0; mt < 8; ++mt) {
      size_t roff = (size_t)(rbase + mt * 16 + L2) * K;
      v16bf ah = loadA(Xhi + roff, k0, hi);
      v16bf al = loadA(Xlo + roff, k0, hi);
      acc[mt] = mm3s(ah, al, bh, bl, acc[mt]);
    }
  }

  if (colok) {
    float bc = bias ? bias[col] : 0.f;
#pragma unroll
    for (int mt = 0; mt < 8; ++mt) {
#pragma unroll
      for (int r = 0; r < 8; ++r) {
        float v = acc[mt][r] + bc;  // D: M = r + 8*hi, N = L2
        size_t idx = (size_t)(rbase + mt * 16 + r + hi * 8) * N + col;
        if (Yf) Yf[idx] = v;
        if (Yhi) {
          __bf16 hb, lb;
          split1(v, hb, lb);
          Yhi[idx] = hb;
          Ylo[idx] = lb;
        }
      }
    }
  }
}

// ---------- attention: per (batch, head), 16 queries x 128 keys, one wave ----------
// All inputs pre-split bf16 hi/lo planes; softmax w/o 1/sqrt(dh) scaling (matches reference).
__global__ void attn_kernel(const __bf16* __restrict__ Qhi, const __bf16* __restrict__ Qlo,
                            const __bf16* __restrict__ Khi, const __bf16* __restrict__ Klo,
                            const __bf16* __restrict__ Vhi, const __bf16* __restrict__ Vlo,
                            float* __restrict__ Out) {
  __shared__ float asmem[16 * 128];
  int b = blockIdx.x, h = blockIdx.y;
  int lane = threadIdx.x & 31;
  int L2 = lane & 15, hi = lane >> 4;

  // hoist Q fragments (shared by all 8 key tiles)
  size_t qoff = (size_t)(b * N_MASKP + L2) * D_MODEL + h * 64;
  v16bf aqh[2], aql[2];
#pragma unroll
  for (int kc = 0; kc < 2; ++kc) {
    aqh[kc] = loadA(Qhi + qoff, kc * 32, hi);
    aql[kc] = loadA(Qlo + qoff, kc * 32, hi);
  }

  v8f acc[8];
  v8f z = {};
#pragma unroll
  for (int t = 0; t < 8; ++t) acc[t] = z;
#pragma unroll
  for (int t = 0; t < 8; ++t) {
    size_t koff = (size_t)(b * S_CURR + t * 16 + L2) * D_MODEL + h * 64;
#pragma unroll
    for (int kc = 0; kc < 2; ++kc) {
      v16bf bh = loadB(Khi + koff, kc * 32, hi);
      v16bf bl = loadB(Klo + koff, kc * 32, hi);
      acc[t] = mm3s(aqh[kc], aql[kc], bh, bl, acc[t]);
    }
  }

  // softmax rows m = r + 8*hi; values across lanes L2=0..15 of this half
#pragma unroll
  for (int r = 0; r < 8; ++r) {
    float mx = -3.4028235e38f;
#pragma unroll
    for (int t = 0; t < 8; ++t) mx = fmaxf(mx, acc[t][r]);
    for (int d = 1; d < 16; d <<= 1) mx = fmaxf(mx, __shfl_xor(mx, d, 32));
    float s = 0.f;
#pragma unroll
    for (int t = 0; t < 8; ++t) {
      float e = __expf(acc[t][r] - mx);
      acc[t][r] = e;
      s += e;
    }
    for (int d = 1; d < 16; d <<= 1) s += __shfl_xor(s, d, 32);
    float inv = 1.0f / s;
#pragma unroll
    for (int t = 0; t < 8; ++t) asmem[(r + hi * 8) * 128 + t * 16 + L2] = acc[t][r] * inv;
  }
  __syncthreads();

  // hoist prob fragments (shared by the 4 output tiles)
  const float* arow = asmem + L2 * 128;
  v16bf aph[4], apl[4];
#pragma unroll
  for (int c = 0; c < 4; ++c) {
    float av[16];
#pragma unroll
    for (int e = 0; e < 16; ++e) av[e] = arow[c * 32 + hi * 16 + e];
    splitf16(av, aph[c], apl[c]);
  }

  // out[16x64] = probs[16x128] @ V_h[128x64]
  v8f oacc[4];
#pragma unroll
  for (int nt = 0; nt < 4; ++nt) oacc[nt] = z;
#pragma unroll
  for (int nt = 0; nt < 4; ++nt) {
#pragma unroll
    for (int c = 0; c < 4; ++c) {
      v16bf bh, bl;
#pragma unroll
      for (int e = 0; e < 16; ++e) {
        size_t ri = (size_t)(b * S_CURR + c * 32 + hi * 16 + e) * D_MODEL + h * 64 + nt * 16 + L2;
        bh[e] = Vhi[ri];
        bl[e] = Vlo[ri];
      }
      oacc[nt] = mm3s(aph[c], apl[c], bh, bl, oacc[nt]);
    }
  }
#pragma unroll
  for (int nt = 0; nt < 4; ++nt)
#pragma unroll
    for (int r = 0; r < 8; ++r)
      Out[(size_t)(b * N_MASKP + r + hi * 8) * D_MODEL + h * 64 + nt * 16 + L2] = oacc[nt][r];
}

// ---------- log-softmax over rows of length 39998 ----------
__global__ void rowmax_kernel(const float* __restrict__ S, float* __restrict__ rmax, int n) {
  __shared__ float red[256];
  int r = blockIdx.x;
  const float* row = S + (size_t)r * n;
  float m = -3.4028235e38f;
  for (int c = threadIdx.x; c < n; c += 256) m = fmaxf(m, row[c]);
  red[threadIdx.x] = m;
  __syncthreads();
  for (int s = 128; s > 0; s >>= 1) {
    if ((int)threadIdx.x < s) red[threadIdx.x] = fmaxf(red[threadIdx.x], red[threadIdx.x + s]);
    __syncthreads();
  }
  if (threadIdx.x == 0) rmax[r] = red[0];
}

__global__ void rowlogsum_kernel(const float* __restrict__ S, const float* __restrict__ rmax,
                                 float* __restrict__ rlog, int n) {
  __shared__ float red[256];
  int r = blockIdx.x;
  const float* row = S + (size_t)r * n;
  float mx = rmax[r];
  float s = 0.f;
  for (int c = threadIdx.x; c < n; c += 256) s += __expf(row[c] - mx);
  red[threadIdx.x] = s;
  __syncthreads();
  for (int t = 128; t > 0; t >>= 1) {
    if ((int)threadIdx.x < t) red[threadIdx.x] += red[threadIdx.x + t];
    __syncthreads();
  }
  if (threadIdx.x == 0) rlog[r] = logf(red[0]);
}

__global__ void lsm_finalize_kernel(float* __restrict__ S, const float* __restrict__ rmax,
                                    const float* __restrict__ rlog, int n) {
  int r = blockIdx.y;
  int c = blockIdx.x * 256 + threadIdx.x;
  if (c < n) {
    size_t i = (size_t)r * n + c;
    S[i] = S[i] - rmax[r] - rlog[r];
  }
}

extern "C" void kernel_launch(void* const* d_in, const int* in_sizes, int n_in,
                              void* d_out, int out_size, void* d_ws, size_t ws_size,
                              hipStream_t stream) {
  (void)in_sizes; (void)n_in; (void)out_size; (void)ws_size;
  // live inputs only (hist path / h_* / x_* / userId are dead code in the reference)
  const int*   mask_pos  = (const int*)d_in[2];
  const int*   mask_curr = (const int*)d_in[3];
  const float* emb       = (const float*)d_in[5];
  const float* c_wq = (const float*)d_in[12];
  const float* c_bq = (const float*)d_in[13];
  const float* c_wk = (const float*)d_in[14];
  const float* c_bk = (const float*)d_in[15];
  const float* c_wv = (const float*)d_in[16];
  const float* c_bv = (const float*)d_in[17];
  const float* t2_w = (const float*)d_in[24];
  const float* t2_b = (const float*)d_in[25];
  char*  w   = (char*)d_ws;
  float* out = (float*)d_out;

  // workspace layout (bytes), ~41 MB total
  float*  curr_f  = (float*)(w + 0);          // 8 MB   32*128*512 fp32
  __bf16* curr_hi = (__bf16*)(w + 8388608);   // 4 MB
  __bf16* curr_lo = (__bf16*)(w + 12582912);  // 4 MB
  __bf16* K_hi    = (__bf16*)(w + 16777216);  // 4 MB
  __bf16* K_lo    = (__bf16*)(w + 20971520);  // 4 MB
  __bf16* V_hi    = (__bf16*)(w + 25165824);  // 4 MB
  __bf16* V_lo    = (__bf16*)(w + 29360128);  // 4 MB
  float*  Qin_f   = (float*)(w + 33554432);   // 1 MB
  __bf16* Qin_hi  = (__bf16*)(w + 34603008);  // 0.5 MB
  __bf16* Qin_lo  = (__bf16*)(w + 35127296);  // 0.5 MB
  __bf16* Q_hi    = (__bf16*)(w + 35651584);  // 0.5 MB
  __bf16* Q_lo    = (__bf16*)(w + 36175872);  // 0.5 MB
  float*  Att_f   = (float*)(w + 36700160);   // 1 MB
  __bf16* Att_hi  = (__bf16*)(w + 37748736);  // 0.5 MB
  __bf16* Att_lo  = (__bf16*)(w + 38273024);  // 0.5 MB
  float*  Gath_f  = (float*)(w + 38797312);   // 1 MB
  __bf16* G_hi    = (__bf16*)(w + 39845888);  // 0.5 MB
  __bf16* G_lo    = (__bf16*)(w + 40370176);  // 0.5 MB
  float*  rmax    = (float*)(w + 40894464);   // 2 KB
  float*  rlog    = (float*)(w + 40896512);   // 2 KB

  // 1) curr = emb[mask_curr] + PE ; split
  build_curr_kernel<<<dim3(S_CURR, B_SZ), 128, 0, stream>>>(mask_curr, emb, curr_f);
  split_kernel<<<dim3(2097152 / 256), 256, 0, stream>>>(curr_f, curr_hi, curr_lo, 2097152, 0);
  // 2) K/V projections (4096x512x512), stored directly as bf16 split planes
  gemm8_kernel<<<dim3(32, 8), 128, 0, stream>>>(curr_hi, curr_lo, c_wk, c_bk,
                                                nullptr, K_hi, K_lo, 4096, 512, 512);
  gemm8_kernel<<<dim3(32, 8), 128, 0, stream>>>(curr_hi, curr_lo, c_wv, c_bv,
                                                nullptr, V_hi, V_lo, 4096, 512, 512);
  // 3) gather 16 masked query rows per batch; project Q, store split
  qgather_kernel<<<dim3(512), 128, 0, stream>>>(mask_pos, curr_f, Qin_f);
  split_kernel<<<dim3(262144 / 256), 256, 0, stream>>>(Qin_f, Qin_hi, Qin_lo, 262144, 0);
  gemm8_kernel<<<dim3(32, 1), 128, 0, stream>>>(Qin_hi, Qin_lo, c_wq, c_bq,
                                                nullptr, Q_hi, Q_lo, 512, 512, 512);
  // 4) attention (16 queries x 128 keys per (b,h))
  attn_kernel<<<dim3(B_SZ, HEADS), 32, 0, stream>>>(Q_hi, Q_lo, K_hi, K_lo, V_hi, V_lo, Att_f);
  // 5) hyb = tanh(att) @ t2_w^T + t2_b (tanh folded into split)
  split_kernel<<<dim3(262144 / 256), 256, 0, stream>>>(Att_f, Att_hi, Att_lo, 262144, 1);
  gemm8_kernel<<<dim3(32, 1), 128, 0, stream>>>(Att_hi, Att_lo, t2_w, t2_b,
                                                Gath_f, nullptr, nullptr, 512, 512, 512);
  split_kernel<<<dim3(262144 / 256), 256, 0, stream>>>(Gath_f, G_hi, G_lo, 262144, 0);
  // 6) scores = gathered @ emb[2:]^T  (512 x 39998) straight into d_out
  gemm8_kernel<<<dim3(2500, 1), 128, 0, stream>>>(G_hi, G_lo, emb + 2 * D_MODEL, nullptr,
                                                  out, nullptr, nullptr, 512, N_CAND, 512);
  // 7) log-softmax rows
  rowmax_kernel<<<dim3(512), 256, 0, stream>>>(out, rmax, N_CAND);
  rowlogsum_kernel<<<dim3(512), 256, 0, stream>>>(out, rmax, rlog, N_CAND);
  lsm_finalize_kernel<<<dim3((N_CAND + 255) / 256, 512), 256, 0, stream>>>(out, rmax, rlog, N_CAND);
}
